// CrossMambaBlock_24412594111023
// MI455X (gfx1250) — matrix operand order repaired and checked by reading, not compiled, not run on hardware
//
#include <hip/hip_runtime.h>
#include <hip/hip_bf16.h>
#include <math.h>

// ---------------- problem constants ----------------
#define DIMC 256
#define LLEN 4096
#define BB 2
#define MROWS (BB * LLEN)        // 8192 token positions
#define NCHUNK 16
#define CHLEN (LLEN / NCHUNK)    // 256

// ---------------- vector types ----------------
typedef __bf16 v16bf __attribute__((ext_vector_type(16)));
typedef float  v8f   __attribute__((ext_vector_type(8)));
typedef unsigned short u16x8  __attribute__((ext_vector_type(8)));
typedef unsigned short u16x16 __attribute__((ext_vector_type(16)));
typedef int b128v __attribute__((vector_size(16)));   // matches builtin param type

// gfx1250 async global->LDS path (ASYNCcnt), guarded for toolchain support
#if __has_builtin(__builtin_amdgcn_global_load_async_to_lds_b128) && \
    __has_builtin(__builtin_amdgcn_s_wait_asynccnt)
#define HAS_ASYNC 1
#else
#define HAS_ASYNC 0
#endif

// ---------------- helpers ----------------
__device__ __forceinline__ unsigned short f2bf(float f) {
  unsigned int u = __float_as_uint(f);
  u += 0x7fffu + ((u >> 16) & 1u);          // round-to-nearest-even
  return (unsigned short)(u >> 16);
}
__device__ __forceinline__ float sigm(float x) { return 1.f / (1.f + __expf(-x)); }
__device__ __forceinline__ float silu_f(float x) { return x * sigm(x); }
__device__ __forceinline__ float softplus_f(float x) {
  return (x > 20.f) ? x : logf(1.f + __expf(x));
}

// 16-byte global->LDS copy: async DMA when available, else vectorized ld/st
__device__ __forceinline__ void copy16_g2l(unsigned short* lds,
                                           const unsigned short* g) {
#if HAS_ASYNC
  __builtin_amdgcn_global_load_async_to_lds_b128(
      (__attribute__((address_space(1))) b128v*)(unsigned short*)g,
      (__attribute__((address_space(3))) b128v*)lds, 0, 0);
#else
  *(u16x8*)lds = *(const u16x8*)g;
#endif
}
__device__ __forceinline__ void async_join() {
#if HAS_ASYNC
  __builtin_amdgcn_s_wait_asynccnt(0);
#endif
}

// block-wide sum over 256 threads (8 wave32), result broadcast to all
__device__ __forceinline__ float block_sum256(float v, float* sm) {
  for (int o = 16; o > 0; o >>= 1) v += __shfl_down(v, o, 32);
  int w = threadIdx.x >> 5, ln = threadIdx.x & 31;
  __syncthreads();
  if (ln == 0) sm[w] = v;
  __syncthreads();
  if (w == 0) {
    float t = (ln < 8) ? sm[ln] : 0.f;
    for (int o = 4; o > 0; o >>= 1) t += __shfl_down(t, o, 32);
    if (ln == 0) sm[0] = t;
  }
  __syncthreads();
  return sm[0];
}

// =====================================================================
// bf16-WMMA GEMM:  C[M,N] = A[M,K] @ W[N,K]^T (+ bias[n])
// A, W pre-converted bf16.  K multiple of 32 (weights zero-padded),
// W buffer rows padded to the 128-tile (zeros), A rows always valid.
// Double-buffered LDS, async global->LDS staging, 8 waves, 128x128 tile.
// =====================================================================
#define TM 128
#define TN 128
#define LDT 40   // padded LDS row stride (32 data + 8 pad) -> conflict-free b128

__global__ __launch_bounds__(256) void gemm_bf16_kernel(
    const unsigned short* __restrict__ A, int lda,
    const unsigned short* __restrict__ W, int ldw,
    const float* __restrict__ bias,
    float* __restrict__ C, unsigned short* __restrict__ Cbf, int ldc,
    int N, int K)
{
  __shared__ __align__(16) unsigned short As[2][TM * LDT];
  __shared__ __align__(16) unsigned short Bs[2][TN * LDT];

  const int tid  = threadIdx.x;
  const int lane = tid & 31;
  const int wave = tid >> 5;
  const int l15  = lane & 15;
  const int hi   = lane >> 4;               // 0 for lanes 0-15, 1 for 16-31
  const int mbase = blockIdx.y * TM;
  const int nbase = blockIdx.x * TN;
  const int wm = (wave >> 1) * 32;          // wave row offset  (0..96)
  const int wn = (wave & 1) * 64;           // wave col offset  (0 or 64)

  auto stage = [&](int buf, int k0) {
    // 128 rows x 32 bf16 = 512 sixteen-byte chunks per tile; 2 per thread
    for (int c = tid; c < (TM * 32) / 8; c += 256) {
      int row = c >> 2, seg = (c & 3) * 8;
      copy16_g2l(&As[buf][row * LDT + seg],
                 A + (size_t)(mbase + row) * lda + k0 + seg);
      copy16_g2l(&Bs[buf][row * LDT + seg],
                 W + (size_t)(nbase + row) * ldw + k0 + seg);
    }
  };

  v8f acc[2][4] = {};
  const int nk = K >> 5;

  stage(0, 0);                                    // prologue: chunk 0 in flight

  for (int i = 0; i < nk; ++i) {
    const int cb = i & 1;
    async_join();                                 // own chunk-i copies done
    __syncthreads();                              // everyone's copies done;
                                                  // everyone's reads of other buf done
    if (i + 1 < nk) stage(cb ^ 1, (i + 1) << 5);  // overlap next chunk with compute

    // ---- fragments per ISA 7.12.2 layouts ----
    u16x16 afr[2], bfr[4];
#pragma unroll
    for (int mi = 0; mi < 2; ++mi) {
      int row = wm + mi * 16 + l15;
      u16x8 lo = *(const u16x8*)&As[cb][row * LDT + hi * 8];       // K {0-7}/{8-15}
      u16x8 up = *(const u16x8*)&As[cb][row * LDT + 16 + hi * 8];  // K {16-23}/{24-31}
      afr[mi] = __builtin_shufflevector(lo, up, 0,1,2,3,4,5,6,7,8,9,10,11,12,13,14,15);
    }
#pragma unroll
    for (int nj = 0; nj < 4; ++nj) {
      int col = wn + nj * 16 + l15;
      u16x8 lo = *(const u16x8*)&Bs[cb][col * LDT + hi * 8];       // K {0-7}/{8-15}
      u16x8 up = *(const u16x8*)&Bs[cb][col * LDT + 16 + hi * 8];  // K {16-23}/{24-31}
      bfr[nj] = __builtin_shufflevector(lo, up, 0,1,2,3,4,5,6,7,8,9,10,11,12,13,14,15);
    }

    // ---- 8 WMMAs per K-chunk ----
#pragma unroll
    for (int mi = 0; mi < 2; ++mi) {
      v16bf a = __builtin_bit_cast(v16bf, afr[mi]);
#pragma unroll
      for (int nj = 0; nj < 4; ++nj) {
        v16bf b = __builtin_bit_cast(v16bf, bfr[nj]);
        acc[mi][nj] = __builtin_amdgcn_wmma_f32_16x16x32_bf16(
            false, a, false, b, (short)0, acc[mi][nj], false, false);
      }
    }
  }

  // ---- store (VGPR r -> rows r / r+8 per half-wave) ----
#pragma unroll
  for (int mi = 0; mi < 2; ++mi) {
#pragma unroll
    for (int nj = 0; nj < 4; ++nj) {
      int mrow = mbase + wm + mi * 16 + hi * 8;
      int col  = nbase + wn + nj * 16 + l15;
      if (col < N) {
        float bv = bias ? bias[col] : 0.f;
#pragma unroll
        for (int r = 0; r < 8; ++r) {
          float v = acc[mi][nj][r] + bv;
          if (C)   C[(size_t)(mrow + r) * ldc + col] = v;
          if (Cbf) Cbf[(size_t)(mrow + r) * ldc + col] = f2bf(v);
        }
      }
    }
  }
}

// =====================================================================
// Weight fp32 -> bf16 with zero padding to (prows, pcols)
// =====================================================================
__global__ __launch_bounds__(256) void wcvt_kernel(
    const float* __restrict__ src, unsigned short* __restrict__ dst,
    int rows, int cols, int prows, int pcols)
{
  int i = blockIdx.x * 256 + threadIdx.x;
  if (i >= prows * pcols) return;
  int r = i / pcols, c = i - r * pcols;
  float v = (r < rows && c < cols) ? src[(size_t)r * cols + c] : 0.f;
  dst[(size_t)r * pcols + c] = f2bf(v);
}

// =====================================================================
// Dual LayerNorm -> fp32 xcat [x0|x1] and bf16 copy for the GEMMs
// =====================================================================
__global__ __launch_bounds__(256) void ln2_kernel(
    const float* __restrict__ in0, const float* __restrict__ in1,
    const float* __restrict__ g0, const float* __restrict__ b0,
    const float* __restrict__ g1, const float* __restrict__ b1,
    float* __restrict__ xcat, unsigned short* __restrict__ xcat_bf)
{
  __shared__ float sm[8];
  const int m = blockIdx.x, d = threadIdx.x;
  float a = in0[(size_t)m * DIMC + d];
  float c = in1[(size_t)m * DIMC + d];
  float s0 = block_sum256(a, sm), q0 = block_sum256(a * a, sm);
  float s1 = block_sum256(c, sm), q1 = block_sum256(c * c, sm);
  float m0 = s0 * (1.f / DIMC), v0 = q0 * (1.f / DIMC) - m0 * m0;
  float m1 = s1 * (1.f / DIMC), v1 = q1 * (1.f / DIMC) - m1 * m1;
  float x0 = (a - m0) * rsqrtf(v0 + 1e-5f) * g0[d] + b0[d];
  float x1 = (c - m1) * rsqrtf(v1 + 1e-5f) * g1[d] + b1[d];
  xcat[(size_t)m * 512 + d]       = x0;
  xcat[(size_t)m * 512 + 256 + d] = x1;
  xcat_bf[(size_t)m * 512 + d]       = f2bf(x0);
  xcat_bf[(size_t)m * 512 + 256 + d] = f2bf(x1);
}

// =====================================================================
// Depthwise causal conv (fwd) + anti-causal reversed-tap conv (bwd) + SiLU
// =====================================================================
__global__ __launch_bounds__(256) void conv_silu_kernel(
    const float* __restrict__ xz,
    const float* __restrict__ cfW, const float* __restrict__ cfb,
    const float* __restrict__ cbW, const float* __restrict__ cbb,
    float* __restrict__ xsf, float* __restrict__ xsb)
{
  const int m = blockIdx.x, d = threadIdx.x;
  const int b = m >> 12, l = m & (LLEN - 1);
  auto xat = [&](int ll) -> float {
    return (ll >= 0 && ll < LLEN) ? xz[(size_t)(b * LLEN + ll) * 512 + d] : 0.f;
  };
  float x0 = xat(l), xm1 = xat(l - 1), xm2 = xat(l - 2), xm3 = xat(l - 3);
  float xp1 = xat(l + 1), xp2 = xat(l + 2), xp3 = xat(l + 3);
  const float* cf = cfW + d * 4;
  const float* cb = cbW + d * 4;
  float f  = cf[0] * xm3 + cf[1] * xm2 + cf[2] * xm1 + cf[3] * x0 + cfb[d];
  float bk = cb[0] * xp3 + cb[1] * xp2 + cb[2] * xp1 + cb[3] * x0 + cbb[d];
  xsf[(size_t)m * DIMC + d] = silu_f(f);
  xsb[(size_t)m * DIMC + d] = silu_f(bk);
}

// =====================================================================
// 3-phase chunked selective scan.  Thread -> (d = dg*16 + t/16, n = t&15).
// =====================================================================
__global__ __launch_bounds__(256) void scan_phase1(
    const float* __restrict__ dtf, const float* __restrict__ dtb,
    const float* __restrict__ xsf, const float* __restrict__ xsb,
    const float* __restrict__ xdf, const float* __restrict__ xdb,
    const float* __restrict__ Alf, const float* __restrict__ Alb,
    float* __restrict__ hend, float* __restrict__ aprod)
{
  const int bx = blockIdx.x;
  const int dg = bx & 15, ck = (bx >> 4) & 15, b = (bx >> 8) & 1, dir = (bx >> 9) & 1;
  const int t = threadIdx.x, n = t & 15, d = dg * 16 + (t >> 4);
  const float* dtpre = dir ? dtb : dtf;
  const float* xs    = dir ? xsb : xsf;
  const float* xd    = dir ? xdb : xdf;
  const float  a     = -__expf((dir ? Alb : Alf)[d * 16 + n]);
  float h = 0.f, sdt = 0.f;
  const int base = b * LLEN;
  for (int i = 0; i < CHLEN; ++i) {
    int ls = ck * CHLEN + i;
    int pos = dir ? (LLEN - 1 - ls) : ls;
    size_t m = (size_t)(base + pos);
    float dt = softplus_f(dtpre[m * DIMC + d]);
    float xv = xs[m * DIMC + d];
    float Bv = xd[m * 48 + 16 + n];
    h = __expf(dt * a) * h + dt * Bv * xv;
    sdt += dt;
  }
  size_t idx = (size_t)((dir * 2 + b) * NCHUNK + ck) * 4096 + d * 16 + n;
  hend[idx]  = h;
  aprod[idx] = __expf(a * sdt);
}

__global__ __launch_bounds__(256) void scan_phase2(
    const float* __restrict__ hend, const float* __restrict__ aprod,
    float* __restrict__ carry)
{
  int gid = blockIdx.x * 256 + threadIdx.x;
  int n = gid & 15, d = (gid >> 4) & 255, b = (gid >> 12) & 1, dir = (gid >> 13) & 1;
  float h = 0.f;
  for (int c = 0; c < NCHUNK; ++c) {
    size_t idx = (size_t)((dir * 2 + b) * NCHUNK + c) * 4096 + d * 16 + n;
    carry[idx] = h;
    h = aprod[idx] * h + hend[idx];
  }
}

__global__ __launch_bounds__(256) void scan_phase3(
    const float* __restrict__ dtf, const float* __restrict__ dtb,
    const float* __restrict__ xsf, const float* __restrict__ xsb,
    const float* __restrict__ xdf, const float* __restrict__ xdb,
    const float* __restrict__ Alf, const float* __restrict__ Alb,
    const float* __restrict__ Df,  const float* __restrict__ Db,
    const float* __restrict__ carry, const float* __restrict__ xz,
    float* __restrict__ ydir)
{
  const int bx = blockIdx.x;
  const int dg = bx & 15, ck = (bx >> 4) & 15, b = (bx >> 8) & 1, dir = (bx >> 9) & 1;
  const int t = threadIdx.x, n = t & 15, d = dg * 16 + (t >> 4);
  const float* dtpre = dir ? dtb : dtf;
  const float* xs    = dir ? xsb : xsf;
  const float* xd    = dir ? xdb : xdf;
  const float  a     = -__expf((dir ? Alb : Alf)[d * 16 + n]);
  const float  Dv    = (dir ? Db : Df)[d];
  size_t sidx = (size_t)((dir * 2 + b) * NCHUNK + ck) * 4096 + d * 16 + n;
  float h = carry[sidx];
  const int base = b * LLEN;
  for (int i = 0; i < CHLEN; ++i) {
    int ls = ck * CHLEN + i;
    int pos = dir ? (LLEN - 1 - ls) : ls;
    size_t m = (size_t)(base + pos);
    float dt = softplus_f(dtpre[m * DIMC + d]);
    float xv = xs[m * DIMC + d];
    float Bv = xd[m * 48 + 16 + n];
    float Cv = xd[m * 48 + 32 + n];
    h = __expf(dt * a) * h + dt * Bv * xv;
    float yn = h * Cv;
    yn += __shfl_xor(yn, 1, 32);
    yn += __shfl_xor(yn, 2, 32);
    yn += __shfl_xor(yn, 4, 32);
    yn += __shfl_xor(yn, 8, 32);
    if (n == 0) {
      float zv = xz[m * 512 + 256 + d];
      ydir[(size_t)dir * MROWS * DIMC + m * DIMC + d] = (yn + Dv * xv) * silu_f(zv);
    }
  }
}

// =====================================================================
// y = 0.5*(yf+yb), out-LayerNorm -> bf16 (feeds out_proj GEMM)
// =====================================================================
__global__ __launch_bounds__(256) void combine_ln_kernel(
    const float* __restrict__ ydir,
    const float* __restrict__ g, const float* __restrict__ bta,
    unsigned short* __restrict__ yln_bf)
{
  __shared__ float sm[8];
  const int m = blockIdx.x, d = threadIdx.x;
  float v = 0.5f * (ydir[(size_t)m * DIMC + d] +
                    ydir[(size_t)MROWS * DIMC + (size_t)m * DIMC + d]);
  float s = block_sum256(v, sm), q = block_sum256(v * v, sm);
  float mu = s * (1.f / DIMC), var = q * (1.f / DIMC) - mu * mu;
  yln_bf[(size_t)m * DIMC + d] =
      f2bf((v - mu) * rsqrtf(var + 1e-5f) * g[d] + bta[d]);
}

// =====================================================================
// Final: post-LN(out_proj) -> gate with sigmoid(w), blend with x0, add skip
// =====================================================================
__global__ __launch_bounds__(256) void final_kernel(
    const float* __restrict__ yproj,
    const float* __restrict__ pg, const float* __restrict__ pb,
    const float* __restrict__ wlog, const float* __restrict__ xcat,
    const float* __restrict__ in0, float* __restrict__ out)
{
  __shared__ float sm[8];
  const int m = blockIdx.x, d = threadIdx.x;
  float v = yproj[(size_t)m * DIMC + d];
  float s = block_sum256(v, sm), q = block_sum256(v * v, sm);
  float mu = s * (1.f / DIMC), var = q * (1.f / DIMC) - mu * mu;
  float pn = (v - mu) * rsqrtf(var + 1e-5f) * pg[d] + pb[d];
  float w  = sigm(wlog[(size_t)m * DIMC + d]);
  float x0 = xcat[(size_t)m * 512 + d];
  out[(size_t)m * DIMC + d] = pn * w + x0 * (1.f - w) + in0[(size_t)m * DIMC + d];
}

// =====================================================================
// Host orchestration
// =====================================================================
extern "C" void kernel_launch(void* const* d_in, const int* in_sizes, int n_in,
                              void* d_out, int out_size, void* d_ws, size_t ws_size,
                              hipStream_t stream) {
  const float* input0  = (const float*)d_in[0];
  const float* input1  = (const float*)d_in[1];
  const float* norm0_g = (const float*)d_in[2];
  const float* norm0_b = (const float*)d_in[3];
  const float* norm1_g = (const float*)d_in[4];
  const float* norm1_b = (const float*)d_in[5];
  const float* cw_W    = (const float*)d_in[6];
  const float* cw_b    = (const float*)d_in[7];
  const float* in_proj_W    = (const float*)d_in[8];
  const float* extra_proj_W = (const float*)d_in[9];
  const float* convf_W = (const float*)d_in[10];
  const float* convf_b = (const float*)d_in[11];
  const float* xprojf_W = (const float*)d_in[12];
  const float* dtf_W   = (const float*)d_in[13];
  const float* dtf_b   = (const float*)d_in[14];
  const float* A_log_f = (const float*)d_in[15];
  const float* D_f     = (const float*)d_in[16];
  const float* convb_W = (const float*)d_in[17];
  const float* convb_b = (const float*)d_in[18];
  const float* xprojb_W = (const float*)d_in[19];
  const float* dtb_W   = (const float*)d_in[20];
  const float* dtb_b   = (const float*)d_in[21];
  const float* A_log_b = (const float*)d_in[22];
  const float* D_b     = (const float*)d_in[23];
  const float* outnorm_g = (const float*)d_in[24];
  const float* outnorm_b = (const float*)d_in[25];
  const float* out_proj_W = (const float*)d_in[26];
  const float* post_g  = (const float*)d_in[27];
  const float* post_b  = (const float*)d_in[28];
  float* out = (float*)d_out;

  // ---- workspace layout (float units); total ~114 MB ----
  const size_t MF = 1048576;
  float* ws = (float*)d_ws;
  size_t o = 0;
  float* xcat = ws + o; o += 4 * MF;     // [M,512] fp32 x0|x1
  float* xz   = ws + o; o += 4 * MF;     // [M,512] x|z
  float* wlog = ws + o; o += 2 * MF;     // [M,256]
  float* dtpf = ws + o; o += 2 * MF;     // [M,256]  (reused as yproj)
  float* dtpb = ws + o; o += 2 * MF;     // [M,256]
  float* xsf  = ws + o; o += 2 * MF;     // [M,256]
  float* xsb  = ws + o; o += 2 * MF;     // [M,256]
  float* xdf  = ws + o; o += MF / 2;     // [M,48]
  float* xdb  = ws + o; o += MF / 2;     // [M,48]
  float* hend = ws + o; o += 262144;
  float* aprod= ws + o; o += 262144;
  float* carry= ws + o; o += 262144;
  float* ydir = ws + o; o += 4 * MF;     // [2,M,256]
  unsigned short* xcat_bf = (unsigned short*)(ws + o); o += 2 * MF;   // M*512 u16
  unsigned short* ep_bf   = (unsigned short*)(ws + o); o += 1 * MF;   // M*256 u16
  unsigned short* yln_bf  = (unsigned short*)(ws + o); o += 1 * MF;   // M*256 u16
  unsigned short* xdf_bf  = (unsigned short*)(ws + o); o += MF / 4;   // M*48  u16
  unsigned short* xdb_bf  = (unsigned short*)(ws + o); o += MF / 4;
  unsigned short* wbf     = (unsigned short*)(ws + o); o += MF / 4;   // weights
  unsigned short* winp = wbf;               // 512x256
  unsigned short* wcw  = wbf + 131072;      // 256x512
  unsigned short* wext = wbf + 262144;      // 256x256
  unsigned short* wxpf = wbf + 327680;      // 128x256 (48 rows valid, rest 0)
  unsigned short* wxpb = wbf + 360448;      // 128x256
  unsigned short* wdtf = wbf + 393216;      // 256x32  (16 cols valid, rest 0)
  unsigned short* wdtb = wbf + 401408;      // 256x32
  unsigned short* wout = wbf + 409600;      // 256x256
  float* yproj = dtpf;

  const dim3 blk(256);
  const int mt = MROWS / TM;   // 64 row tiles

  // 0) weight conversion (fp32 -> bf16, zero-padded)
  wcvt_kernel<<<512, blk, 0, stream>>>(in_proj_W,    winp, 512, 256, 512, 256);
  wcvt_kernel<<<512, blk, 0, stream>>>(cw_W,         wcw,  256, 512, 256, 512);
  wcvt_kernel<<<256, blk, 0, stream>>>(extra_proj_W, wext, 256, 256, 256, 256);
  wcvt_kernel<<<128, blk, 0, stream>>>(xprojf_W,     wxpf,  48, 256, 128, 256);
  wcvt_kernel<<<128, blk, 0, stream>>>(xprojb_W,     wxpb,  48, 256, 128, 256);
  wcvt_kernel<<<32,  blk, 0, stream>>>(dtf_W,        wdtf, 256,  16, 256,  32);
  wcvt_kernel<<<32,  blk, 0, stream>>>(dtb_W,        wdtb, 256,  16, 256,  32);
  wcvt_kernel<<<256, blk, 0, stream>>>(out_proj_W,   wout, 256, 256, 256, 256);

  // 1) dual LayerNorm -> xcat (fp32 + bf16)
  ln2_kernel<<<MROWS, blk, 0, stream>>>(input0, input1, norm0_g, norm0_b,
                                        norm1_g, norm1_b, xcat, xcat_bf);
  // 2) xz = x0 @ in_proj^T  (N=512,K=256)
  gemm_bf16_kernel<<<dim3(4, mt), blk, 0, stream>>>(
      xcat_bf, 512, winp, 256, nullptr, xz, nullptr, 512, 512, 256);
  // 3) wlog = [x0,x1] @ cw^T + cw_b  (N=256,K=512)
  gemm_bf16_kernel<<<dim3(2, mt), blk, 0, stream>>>(
      xcat_bf, 512, wcw, 512, cw_b, wlog, nullptr, 256, 256, 512);
  // 4) ep = x1 @ extra_proj^T  (bf16-only output; feeds xproj GEMMs)
  gemm_bf16_kernel<<<dim3(2, mt), blk, 0, stream>>>(
      xcat_bf + 256, 512, wext, 256, nullptr, nullptr, ep_bf, 256, 256, 256);
  // 5) xdbl_{f,b} = ep @ xproj^T  (N=48,K=256; fp32 for scan + bf16 for dt GEMM)
  gemm_bf16_kernel<<<dim3(1, mt), blk, 0, stream>>>(
      ep_bf, 256, wxpf, 256, nullptr, xdf, xdf_bf, 48, 48, 256);
  gemm_bf16_kernel<<<dim3(1, mt), blk, 0, stream>>>(
      ep_bf, 256, wxpb, 256, nullptr, xdb, xdb_bf, 48, 48, 256);
  // 6) dtpre_{f,b} = xdbl[:, :16] @ dt_W^T + dt_b  (K padded 16->32 with 0 weights)
  gemm_bf16_kernel<<<dim3(2, mt), blk, 0, stream>>>(
      xdf_bf, 48, wdtf, 32, dtf_b, dtpf, nullptr, 256, 256, 32);
  gemm_bf16_kernel<<<dim3(2, mt), blk, 0, stream>>>(
      xdb_bf, 48, wdtb, 32, dtb_b, dtpb, nullptr, 256, 256, 32);
  // 7) depthwise convs + SiLU
  conv_silu_kernel<<<MROWS, blk, 0, stream>>>(xz, convf_W, convf_b,
                                              convb_W, convb_b, xsf, xsb);
  // 8) chunked selective scan (both directions)
  scan_phase1<<<1024, blk, 0, stream>>>(dtpf, dtpb, xsf, xsb, xdf, xdb,
                                        A_log_f, A_log_b, hend, aprod);
  scan_phase2<<<64, blk, 0, stream>>>(hend, aprod, carry);
  scan_phase3<<<1024, blk, 0, stream>>>(dtpf, dtpb, xsf, xsb, xdf, xdb,
                                        A_log_f, A_log_b, D_f, D_b,
                                        carry, xz, ydir);
  // 9) combine + out-norm -> bf16
  combine_ln_kernel<<<MROWS, blk, 0, stream>>>(ydir, outnorm_g, outnorm_b, yln_bf);
  // 10) out_proj (N=256,K=256)
  gemm_bf16_kernel<<<dim3(2, mt), blk, 0, stream>>>(
      yln_bf, 256, wout, 256, nullptr, yproj, nullptr, 256, 256, 256);
  // 11) post-LN + gated blend + skip
  final_kernel<<<MROWS, blk, 0, stream>>>(yproj, post_g, post_b, wlog,
                                          xcat, input0, out);
}
